// Transformer_39230231282370
// MI455X (gfx1250) — matrix operand order repaired and checked by reading, not compiled, not run on hardware
//
#include <hip/hip_runtime.h>
#include <math.h>

// Problem constants (match reference)
constexpr int CB = 4, CN = 512, CD = 1024, CH = 16, CDK = 64, CDI = 4096, CL = 6, CC = 16000;

typedef __attribute__((ext_vector_type(16))) __bf16 bf16x16;
typedef __attribute__((ext_vector_type(8)))  __bf16 bf16x8;
typedef __attribute__((ext_vector_type(4)))  __bf16 bf16x4;
typedef __attribute__((ext_vector_type(8)))  float  f32x8;

#define TM 256
#define TN 64
#define TK 64
#define APITCH (TK + 8)   // 72 bf16 = 144B: keeps 16B row alignment, staggers banks
#define BPITCH (TK + 8)

static __device__ __forceinline__ bf16x4 cvt4(float4 f) {
  bf16x4 r;
  r.x = (__bf16)f.x; r.y = (__bf16)f.y; r.z = (__bf16)f.z; r.w = (__bf16)f.w;
  return r;
}

// ---------------------------------------------------------------------------
// Batched GEMM: C = alpha * A*B (+bias) (+residual), optional ReLU.
// A: [M,K] row-major (lda). B: [K,N] row-major (ldb) or [N,K] if transB.
// REQUIRES (true for every call in this network): M%256==0, N%64==0, K%64==0,
// lda/ldb/ldc multiples of 4, 16B-aligned base pointers.
// Batch z = zb*Hdiv + zh with two-component strides per operand.
// 8 waves; wave w computes rows [32w,32w+32) x 64 cols: 2x4 WMMA tiles ->
// 8 independent accumulation chains, 16 v_wmma_f32_16x16x32_bf16 per
// K-iteration (TK=64). Register-prefetch double buffering overlaps global
// loads with the WMMA chain.
// ---------------------------------------------------------------------------
__global__ __launch_bounds__(256) void gemm_bf16_wmma(
    const float* __restrict__ A, const float* __restrict__ Bm,
    const float* __restrict__ bias, const float* __restrict__ resid,
    float* __restrict__ Cm,
    int M, int N, int K, int lda, int ldb, int ldc,
    long long sAb, long long sAh, long long sBb, long long sBh,
    long long sCb, long long sCh, int Hdiv,
    float alpha, int relu, int transB)
{
  __shared__ __bf16 As[TM][APITCH];
  __shared__ __bf16 Bs[TN][BPITCH];   // Bs[n][k]: k-contiguous per output column

  const int z  = blockIdx.z;
  const int zb = z / Hdiv, zh = z % Hdiv;
  const float* Ab = A  + zb * sAb + zh * sAh;
  const float* Bb = Bm + zb * sBb + zh * sBh;
  float*       Cb = Cm + zb * sCb + zh * sCh;
  const float* Rb = resid ? (resid + zb * sCb + zh * sCh) : nullptr;

  const int m0 = blockIdx.y * TM;
  const int n0 = blockIdx.x * TN;
  const int tid  = threadIdx.x;
  const int wave = tid >> 5;         // 0..7 -> 32-row slab
  const int lane = tid & 31;
  const int half = lane >> 4;        // K-half selector per ISA layout
  const int lm   = lane & 15;

  // Staging thread mappings (exact, no remainders)
  const int ar  = tid >> 4;          // A: 16 rows per pass, 16 passes
  const int ac4 = (tid & 15) * 4;    // A: float4 column
  const int kb  = tid >> 4;          // B(NN): 4-k block row
  const int nb  = tid & 15;          // B(NN): 4-n block col
  const int bn  = tid & 63;          // B(NT): n row
  const int bk4 = tid >> 6;          // B(NT): float4 k slot (0..3)

  f32x8 acc[2][4] = {};
  float4 Areg[16];
  float4 Breg[4];

  // ---- prefetch tile 0 ----
  {
    #pragma unroll
    for (int it = 0; it < 16; ++it)
      Areg[it] = *(const float4*)&Ab[(long long)(m0 + ar + it * 16) * lda + ac4];
    if (transB) {
      #pragma unroll
      for (int i = 0; i < 4; ++i)
        Breg[i] = *(const float4*)&Bb[(long long)(n0 + bn) * ldb + (bk4 + i * 4) * 4];
    } else {
      #pragma unroll
      for (int i = 0; i < 4; ++i)
        Breg[i] = *(const float4*)&Bb[(long long)(kb * 4 + i) * ldb + n0 + nb * 4];
    }
  }

  for (int k0 = 0; k0 < K; k0 += TK) {
    __syncthreads();   // previous compute done reading LDS

    // ---- store staged regs to LDS (fp32 -> bf16) ----
    #pragma unroll
    for (int it = 0; it < 16; ++it)
      *(bf16x4*)&As[ar + it * 16][ac4] = cvt4(Areg[it]);
    if (transB) {
      #pragma unroll
      for (int i = 0; i < 4; ++i)
        *(bf16x4*)&Bs[bn][(bk4 + i * 4) * 4] = cvt4(Breg[i]);
    } else {
      // 4x4 register transpose: Breg[i] holds k=kb*4+i, n=nb*4..+3
      #pragma unroll
      for (int j = 0; j < 4; ++j) {
        bf16x4 t;
        t.x = (__bf16)((const float*)&Breg[0])[j];
        t.y = (__bf16)((const float*)&Breg[1])[j];
        t.z = (__bf16)((const float*)&Breg[2])[j];
        t.w = (__bf16)((const float*)&Breg[3])[j];
        *(bf16x4*)&Bs[nb * 4 + j][kb * 4] = t;
      }
    }
    __syncthreads();   // LDS tile ready

    // ---- issue next tile's global loads (overlap with WMMA below) ----
    int kn = k0 + TK;
    if (kn < K) {
      #pragma unroll
      for (int it = 0; it < 16; ++it)
        Areg[it] = *(const float4*)&Ab[(long long)(m0 + ar + it * 16) * lda + kn + ac4];
      if (transB) {
        #pragma unroll
        for (int i = 0; i < 4; ++i)
          Breg[i] = *(const float4*)&Bb[(long long)(n0 + bn) * ldb + kn + (bk4 + i * 4) * 4];
      } else {
        #pragma unroll
        for (int i = 0; i < 4; ++i)
          Breg[i] = *(const float4*)&Bb[(long long)(kn + kb * 4 + i) * ldb + n0 + nb * 4];
      }
    }

    // ---- compute: 2 K-subtiles of 32; 2x4 WMMA tiles each ----
    #pragma unroll
    for (int ks = 0; ks < 2; ++ks) {
      const int kofs = ks * 32;
      // B frag: col = lane%16; lane<16 -> K{0..7,16..23}, lane>=16 -> K{8..15,24..31}
      bf16x16 bf[4];
      #pragma unroll
      for (int j = 0; j < 4; ++j) {
        bf16x8 b0 = *(const bf16x8*)&Bs[j * 16 + lm][kofs + 8 * half];
        bf16x8 b1 = *(const bf16x8*)&Bs[j * 16 + lm][kofs + 16 + 8 * half];
        bf[j] = __builtin_shufflevector(b0, b1, 0,1,2,3,4,5,6,7,8,9,10,11,12,13,14,15);
      }
      #pragma unroll
      for (int i = 0; i < 2; ++i) {
        // A frag: row = lane%16; lane<16 -> K{0..7,16..23}, lane>=16 -> K{8..15,24..31}
        const __bf16* arow = &As[wave * 32 + i * 16 + lm][kofs];
        bf16x8 a0 = *(const bf16x8*)&arow[8 * half];
        bf16x8 a1 = *(const bf16x8*)&arow[16 + 8 * half];
        bf16x16 a = __builtin_shufflevector(a0, a1, 0,1,2,3,4,5,6,7,8,9,10,11,12,13,14,15);
        #pragma unroll
        for (int j = 0; j < 4; ++j)
          acc[i][j] = __builtin_amdgcn_wmma_f32_16x16x32_bf16(
              false, a, false, bf[j], (short)0, acc[i][j], false, false);
      }
    }
  }

  // ---- epilogue: row = base + r + 8*(lane>=16), col = base + lane%16 ----
  #pragma unroll
  for (int i = 0; i < 2; ++i) {
    #pragma unroll
    for (int j = 0; j < 4; ++j) {
      int col = n0 + j * 16 + lm;
      int row_base = m0 + wave * 32 + i * 16 + 8 * half;
      float bv = bias ? bias[col] : 0.0f;
      #pragma unroll
      for (int r = 0; r < 8; ++r) {
        int row = row_base + r;
        float v = alpha * acc[i][j][r] + bv;
        if (Rb)   v += Rb[(long long)row * ldc + col];
        if (relu) v = v > 0.0f ? v : 0.0f;
        Cb[(long long)row * ldc + col] = v;
      }
    }
  }
}

// ---------------------------------------------------------------------------
// x + sinusoidal positional encoding
// ---------------------------------------------------------------------------
__global__ __launch_bounds__(256) void add_pe_kernel(
    const float* __restrict__ x, float* __restrict__ out, int Nn, int Dd, long long total)
{
  long long idx = (long long)blockIdx.x * 256 + threadIdx.x;
  if (idx >= total) return;
  int d = (int)(idx % Dd);
  long long bn = idx / Dd;
  int n = (int)(bn % Nn);
  float div = expf(-(float)(d & ~1) * (logf(10000.0f) / (float)Dd));
  float ang = (float)n * div;
  float pe  = (d & 1) ? cosf(ang) : sinf(ang);
  out[idx] = x[idx] + pe;
}

// ---------------------------------------------------------------------------
// LayerNorm over last dim (block per row)
// ---------------------------------------------------------------------------
__global__ __launch_bounds__(256) void layernorm_kernel(
    const float* __restrict__ x, const float* __restrict__ g, const float* __restrict__ b,
    float* __restrict__ out, int Dd)
{
  __shared__ float red[256];
  long long row = blockIdx.x;
  const float* xr = x + row * (long long)Dd;
  float* orow = out + row * (long long)Dd;
  int tid = threadIdx.x;

  float s = 0.0f;
  for (int i = tid; i < Dd; i += 256) s += xr[i];
  red[tid] = s; __syncthreads();
  for (int off = 128; off; off >>= 1) { if (tid < off) red[tid] += red[tid + off]; __syncthreads(); }
  float mu = red[0] / (float)Dd;
  __syncthreads();

  float v = 0.0f;
  for (int i = tid; i < Dd; i += 256) { float d0 = xr[i] - mu; v += d0 * d0; }
  red[tid] = v; __syncthreads();
  for (int off = 128; off; off >>= 1) { if (tid < off) red[tid] += red[tid + off]; __syncthreads(); }
  float rstd = rsqrtf(red[0] / (float)Dd + 1e-5f);
  __syncthreads();

  for (int i = tid; i < Dd; i += 256)
    orow[i] = (xr[i] - mu) * rstd * g[i] + b[i];
}

// ---------------------------------------------------------------------------
// Masked softmax over attention rows. s: [B,H,N,N]; mask: [B,N,N].
// One block per (b,h,n) row.
// ---------------------------------------------------------------------------
__global__ __launch_bounds__(256) void softmax_mask_kernel(
    float* __restrict__ s, const int* __restrict__ mask, int Hh, int Ncols)
{
  __shared__ float red[256];
  long long row = blockIdx.x;
  int tid = threadIdx.x;
  int n = (int)(row % Ncols);
  long long z = row / Ncols;       // b*H + h
  int b = (int)(z / Hh);
  float* sr = s + row * (long long)Ncols;
  const int* mr = mask + ((long long)b * Ncols + n) * (long long)Ncols;

  float mx = -INFINITY;
  for (int i = tid; i < Ncols; i += 256)
    if (mr[i]) mx = fmaxf(mx, sr[i]);
  red[tid] = mx; __syncthreads();
  for (int off = 128; off; off >>= 1) { if (tid < off) red[tid] = fmaxf(red[tid], red[tid + off]); __syncthreads(); }
  mx = red[0]; __syncthreads();

  float sum = 0.0f;
  for (int i = tid; i < Ncols; i += 256) {
    float e = mr[i] ? expf(sr[i] - mx) : 0.0f;
    sr[i] = e;
    sum += e;
  }
  red[tid] = sum; __syncthreads();
  for (int off = 128; off; off >>= 1) { if (tid < off) red[tid] += red[tid + off]; __syncthreads(); }
  float inv = 1.0f / red[0];
  __syncthreads();
  for (int i = tid; i < Ncols; i += 256) sr[i] *= inv;
}

// ---------------------------------------------------------------------------
// Plain row softmax (final prediction head)
// ---------------------------------------------------------------------------
__global__ __launch_bounds__(256) void softmax_rows_kernel(float* __restrict__ x, int Ncols)
{
  __shared__ float red[256];
  long long row = blockIdx.x;
  int tid = threadIdx.x;
  float* xr = x + row * (long long)Ncols;

  float mx = -INFINITY;
  for (int i = tid; i < Ncols; i += 256) mx = fmaxf(mx, xr[i]);
  red[tid] = mx; __syncthreads();
  for (int off = 128; off; off >>= 1) { if (tid < off) red[tid] = fmaxf(red[tid], red[tid + off]); __syncthreads(); }
  mx = red[0]; __syncthreads();

  float sum = 0.0f;
  for (int i = tid; i < Ncols; i += 256) { float e = expf(xr[i] - mx); xr[i] = e; sum += e; }
  red[tid] = sum; __syncthreads();
  for (int off = 128; off; off >>= 1) { if (tid < off) red[tid] += red[tid + off]; __syncthreads(); }
  float inv = 1.0f / red[0];
  __syncthreads();
  for (int i = tid; i < Ncols; i += 256) xr[i] *= inv;
}

// ---------------------------------------------------------------------------
// Host-side launch helpers
// ---------------------------------------------------------------------------
typedef long long ll;

static inline void gemm(hipStream_t st, const float* A, const float* B,
                        const float* bias, const float* resid, float* C,
                        int M, int N, int K, int lda, int ldb, int ldc,
                        ll sAb, ll sAh, ll sBb, ll sBh, ll sCb, ll sCh,
                        int Hdiv, int batches, float alpha, int relu, int transB)
{
  dim3 grid((unsigned)(N / TN), (unsigned)(M / TM), (unsigned)batches);
  gemm_bf16_wmma<<<grid, 256, 0, st>>>(A, B, bias, resid, C, M, N, K, lda, ldb, ldc,
                                       sAb, sAh, sBb, sBh, sCb, sCh, Hdiv,
                                       alpha, relu, transB);
}

// Multi-head attention block: out = resid + MHA(q_in, kv_in, kv_in)
// q,k,v stored [B,H,N,DK]; s [B,H,N,N]; cat [B,N,H*DK]; out [B,N,D]
static void mha_block(hipStream_t st, const float* q_in, const float* kv_in,
                      const float* Wq, const float* Wk, const float* Wv, const float* Wo,
                      const int* mask, const float* resid,
                      float* q, float* k, float* v, float* s, float* cat, float* out)
{
  const ll sX   = (ll)CN * CD;          // per-batch stride of [B,N,D] activations
  const ll sW   = (ll)CD * CDK;         // per-head stride of Wq/Wk/Wv
  const ll sHD  = (ll)CN * CDK;         // per-head stride of q/k/v
  const ll sBH  = (ll)CH * sHD;         // per-batch stride of q/k/v
  const ll sS_h = (ll)CN * CN;
  const ll sS_b = (ll)CH * sS_h;

  // Projections: batch over (b,h); M=N, N=DK, K=D
  gemm(st, q_in,  Wq, nullptr, nullptr, q, CN, CDK, CD, CD, CDK, CDK,
       sX, 0, 0, sW, sBH, sHD, CH, CB * CH, 1.0f, 0, 0);
  gemm(st, kv_in, Wk, nullptr, nullptr, k, CN, CDK, CD, CD, CDK, CDK,
       sX, 0, 0, sW, sBH, sHD, CH, CB * CH, 1.0f, 0, 0);
  gemm(st, kv_in, Wv, nullptr, nullptr, v, CN, CDK, CD, CD, CDK, CDK,
       sX, 0, 0, sW, sBH, sHD, CH, CB * CH, 1.0f, 0, 0);

  // Scores: s = q @ k^T / sqrt(DK)   (NT GEMM), M=N=512, K=DK
  gemm(st, q, k, nullptr, nullptr, s, CN, CN, CDK, CDK, CDK, CN,
       sBH, sHD, sBH, sHD, sS_b, sS_h, CH, CB * CH, 1.0f / 8.0f, 0, 1);

  // Masked softmax over each (b,h,n) row
  softmax_mask_kernel<<<(unsigned)(CB * CH * CN), 256, 0, st>>>(s, mask, CH, CN);

  // o = a @ v, scattered into concatenated layout cat[b,n,h*DK+dk]
  gemm(st, s, v, nullptr, nullptr, cat, CN, CDK, CN, CN, CDK, CD,
       sS_b, sS_h, sBH, sHD, (ll)CN * CD, (ll)CDK, CH, CB * CH, 1.0f, 0, 0);

  // Output projection with residual: out = cat @ Wo + resid
  gemm(st, cat, Wo, nullptr, resid, out, CB * CN, CD, CD, CD, CD, CD,
       0, 0, 0, 0, 0, 0, 1, 1, 1.0f, 0, 0);
}

extern "C" void kernel_launch(void* const* d_in, const int* in_sizes, int n_in,
                              void* d_out, int out_size, void* d_ws, size_t ws_size,
                              hipStream_t stream)
{
  // ---- inputs (setup_inputs dict order) ----
  const float* enc_in = (const float*)d_in[0];
  const float* dec_in = (const float*)d_in[1];
  const int*   emask  = (const int*)d_in[2];
  const int*   dmask  = (const int*)d_in[3];
  const float* eWq = (const float*)d_in[4];
  const float* eWk = (const float*)d_in[5];
  const float* eWv = (const float*)d_in[6];
  const float* eWo = (const float*)d_in[7];
  const float* eW1 = (const float*)d_in[8];
  const float* eb1 = (const float*)d_in[9];
  const float* eW2 = (const float*)d_in[10];
  const float* eb2 = (const float*)d_in[11];
  const float* eg1 = (const float*)d_in[12];
  const float* ebt1= (const float*)d_in[13];
  const float* eg2 = (const float*)d_in[14];
  const float* ebt2= (const float*)d_in[15];
  const float* sWq = (const float*)d_in[16];
  const float* sWk = (const float*)d_in[17];
  const float* sWv = (const float*)d_in[18];
  const float* sWo = (const float*)d_in[19];
  const float* cWq = (const float*)d_in[20];
  const float* cWk = (const float*)d_in[21];
  const float* cWv = (const float*)d_in[22];
  const float* cWo = (const float*)d_in[23];
  const float* dW1 = (const float*)d_in[24];
  const float* db1 = (const float*)d_in[25];
  const float* dW2 = (const float*)d_in[26];
  const float* db2 = (const float*)d_in[27];
  const float* dg1 = (const float*)d_in[28];
  const float* dbt1= (const float*)d_in[29];
  const float* dg2 = (const float*)d_in[30];
  const float* dbt2= (const float*)d_in[31];
  const float* dg3 = (const float*)d_in[32];
  const float* dbt3= (const float*)d_in[33];
  const float* pW  = (const float*)d_in[34];
  const float* pb  = (const float*)d_in[35];

  // ---- workspace layout (floats) ----
  float* ws  = (float*)d_ws;
  float* xe  = ws;                          // [B,N,D]
  float* xd  = xe  + (ll)CB * CN * CD;      // [B,N,D]
  float* q   = xd  + (ll)CB * CN * CD;      // [B,H,N,DK]
  float* k   = q   + (ll)CB * CH * CN * CDK;
  float* v   = k   + (ll)CB * CH * CN * CDK;
  float* cat = v   + (ll)CB * CH * CN * CDK; // [B,N,H*DK]
  float* t1  = cat + (ll)CB * CN * CD;       // [B,N,D]
  float* hid = t1  + (ll)CB * CN * CD;       // [B,N,DI]
  float* s   = hid + (ll)CB * CN * CDI;      // [B,H,N,N]

  const ll total = (ll)CB * CN * CD;
  const ll wqkvL = (ll)CH * CD * CDK;   // per-layer stride of Wq/Wk/Wv
  const ll woL   = (ll)CH * CDK * CD;   // per-layer stride of Wo
  const ll w1L   = (ll)CD * CDI;
  const ll w2L   = (ll)CDI * CD;

  // ================= Encoder =================
  add_pe_kernel<<<(unsigned)((total + 255) / 256), 256, 0, stream>>>(enc_in, xe, CN, CD, total);

  for (int l = 0; l < CL; ++l) {
    mha_block(stream, xe, xe,
              eWq + (ll)l * wqkvL, eWk + (ll)l * wqkvL, eWv + (ll)l * wqkvL,
              eWo + (ll)l * woL, emask, xe, q, k, v, s, cat, t1);
    layernorm_kernel<<<(unsigned)(CB * CN), 256, 0, stream>>>(t1, eg1 + (ll)l * CD, ebt1 + (ll)l * CD, xe, CD);

    gemm(stream, xe, eW1 + (ll)l * w1L, eb1 + (ll)l * CDI, nullptr, hid,
         CB * CN, CDI, CD, CD, CDI, CDI, 0, 0, 0, 0, 0, 0, 1, 1, 1.0f, /*relu*/1, 0);
    gemm(stream, hid, eW2 + (ll)l * w2L, eb2 + (ll)l * CD, /*resid*/xe, t1,
         CB * CN, CD, CDI, CDI, CD, CD, 0, 0, 0, 0, 0, 0, 1, 1, 1.0f, 0, 0);
    layernorm_kernel<<<(unsigned)(CB * CN), 256, 0, stream>>>(t1, eg2 + (ll)l * CD, ebt2 + (ll)l * CD, xe, CD);
  }
  // xe now holds enc_out

  // ================= Decoder =================
  add_pe_kernel<<<(unsigned)((total + 255) / 256), 256, 0, stream>>>(dec_in, xd, CN, CD, total);

  for (int l = 0; l < CL; ++l) {
    // self-attention (causal mask)
    mha_block(stream, xd, xd,
              sWq + (ll)l * wqkvL, sWk + (ll)l * wqkvL, sWv + (ll)l * wqkvL,
              sWo + (ll)l * woL, dmask, xd, q, k, v, s, cat, t1);
    layernorm_kernel<<<(unsigned)(CB * CN), 256, 0, stream>>>(t1, dg1 + (ll)l * CD, dbt1 + (ll)l * CD, xd, CD);

    // cross-attention (q from decoder state, k/v from encoder output)
    mha_block(stream, xd, xe,
              cWq + (ll)l * wqkvL, cWk + (ll)l * wqkvL, cWv + (ll)l * wqkvL,
              cWo + (ll)l * woL, emask, xd, q, k, v, s, cat, t1);
    layernorm_kernel<<<(unsigned)(CB * CN), 256, 0, stream>>>(t1, dg2 + (ll)l * CD, dbt2 + (ll)l * CD, xd, CD);

    // FFN
    gemm(stream, xd, dW1 + (ll)l * w1L, db1 + (ll)l * CDI, nullptr, hid,
         CB * CN, CDI, CD, CD, CDI, CDI, 0, 0, 0, 0, 0, 0, 1, 1, 1.0f, 1, 0);
    gemm(stream, hid, dW2 + (ll)l * w2L, db2 + (ll)l * CD, /*resid*/xd, t1,
         CB * CN, CD, CDI, CDI, CD, CD, 0, 0, 0, 0, 0, 0, 1, 1, 1.0f, 0, 0);
    layernorm_kernel<<<(unsigned)(CB * CN), 256, 0, stream>>>(t1, dg3 + (ll)l * CD, dbt3 + (ll)l * CD, xd, CD);
  }

  // ================= Prediction head =================
  float* out = (float*)d_out;   // [B*N, C]
  gemm(stream, xd, pW, pb, nullptr, out, CB * CN, CC, CD, CD, CC, CC,
       0, 0, 0, 0, 0, 0, 1, 1, 1.0f, 0, 0);
  softmax_rows_kernel<<<(unsigned)(CB * CN), 256, 0, stream>>>(out, CC);
}
